// Clone_Detection_73976516706558
// MI455X (gfx1250) — compile-verified, matrix operand order checked by reading, not run on hardware
//
#include <hip/hip_runtime.h>
#include <hip/hip_bf16.h>

typedef __attribute__((ext_vector_type(16))) _Float16 v16h;
typedef __attribute__((ext_vector_type(8)))  float    v8f;

// ---------------------------------------------------------------------------
// WMMA helpers (CDNA5 gfx1250, wave32).
// A fragment (16x32 f16): lane L: row M = L&15, hi = L>>4.
//   element e -> K = (e/8)*16 + hi*8 + (e&7)          (ISA 7.12.2 16-bit A)
// B fragment (32x16 f16): symmetric, lane -> col N = L&15, same K map.
// C/D (16x16 f32, 8 VGPRs): VGPR r -> row M = r + 8*hi, col N = lane&15.
// ---------------------------------------------------------------------------
__device__ __forceinline__ int kmap(int e, int hi) {
  return ((e >> 3) << 4) + (hi << 3) + (e & 7);
}

__device__ __forceinline__ v8f wmma_f16(v16h a, v16h b, v8f c) {
  return __builtin_amdgcn_wmma_f32_16x16x32_f16(
      /*neg_a=*/false, a, /*neg_b=*/false, b,
      /*c_mod=*/(short)0, c, /*reuse_a=*/false, /*reuse_b=*/false);
}

// Load one 16x32 K-slice of A from f32 row-major X (leading dim = kin),
// zero-padding K >= kin, with optional per-row scale.
__device__ __forceinline__ v16h load_a(const float* __restrict__ X, int row,
                                       int kin, int hi, int slice, float scale) {
  v16h a;
  const float* p = X + (size_t)row * kin;
#pragma unroll
  for (int e = 0; e < 16; ++e) {
    int k = slice * 32 + kmap(e, hi);
    float v = (k < kin) ? p[k] * scale : 0.0f;
    a[e] = (_Float16)v;
  }
  return a;
}

// Load one 32x16 K-slice of B from weight W [koutTotal x kin] row-major
// (computes X @ W^T): B[k][n] = W[colBase+n][k].
__device__ __forceinline__ v16h load_b(const float* __restrict__ W, int colBase,
                                       int col, int kin, int hi, int slice) {
  v16h b;
  int c = colBase + col;
#pragma unroll
  for (int e = 0; e < 16; ++e) {
    int k = slice * 32 + kmap(e, hi);
    float v = (k < kin) ? W[(size_t)c * kin + k] : 0.0f;
    b[e] = (_Float16)v;
  }
  return b;
}

// Scalar (SGPR) wave index so tile range checks are s_cmp/s_cbranch and the
// row-base address math stays scalar.
__device__ __forceinline__ int scalar_wave_id() {
  int wib = __builtin_amdgcn_readfirstlane((int)(threadIdx.x >> 5));
  return (int)blockIdx.x * (int)(blockDim.x >> 5) + wib;
}

// ---------------------------------------------------------------------------
// Generic dense layer: Y = relu(X @ W^T + bias), one 16-row tile per wave.
// SCALE: multiply row of X by rsqrt(max(deg[row],1))  (GCN norm).
// READOUT: instead of storing Y, atomic-add into gout[nid[row]*KOUT+col].
// RAGGED: compiled-in bounds handling; host launches RAGGED=false when
//         nrows % 16 == 0 (the actual case), giving a branch-free store path.
// ---------------------------------------------------------------------------
template <int KIN, int KOUT, bool SCALE, bool READOUT, bool RAGGED>
__global__ void dense_layer(const float* __restrict__ X,
                            const float* __restrict__ W,
                            const float* __restrict__ bias,
                            float* __restrict__ Y,
                            const float* __restrict__ deg,
                            const int* __restrict__ nid,
                            float* __restrict__ gout, int nrows) {
  int r0 = scalar_wave_id() * 16;
  if (r0 >= nrows) return;  // scalar, wave-uniform: EXEC stays all-ones
  int lane = threadIdx.x & 31;
  int sub = lane & 15;
  int hi = lane >> 4;
  int rowA = r0 + sub;
  if (RAGGED && rowA >= nrows) rowA = nrows - 1;
  float scale = 1.0f;
  if (SCALE) scale = rsqrtf(fmaxf(deg[rowA], 1.0f));

  v16h a0 = load_a(X, rowA, KIN, hi, 0, scale);
  v16h a1{};
  if (KIN > 32) a1 = load_a(X, rowA, KIN, hi, 1, scale);

#pragma unroll
  for (int ot = 0; ot < KOUT / 16; ++ot) {
    v8f c = {};
    v16h b0 = load_b(W, ot * 16, sub, KIN, hi, 0);
    c = wmma_f16(a0, b0, c);
    if (KIN > 32) {
      v16h b1 = load_b(W, ot * 16, sub, KIN, hi, 1);
      c = wmma_f16(a1, b1, c);
    }
    int col = ot * 16 + sub;
    float bv = bias[col];
    // base row for this lane's C slots: r0 + 8*hi; slot r adds r rows.
    float* ybase = Y + (size_t)(r0 + 8 * hi) * KOUT + col;
#pragma unroll
    for (int r = 0; r < 8; ++r) {
      int row = r0 + r + 8 * hi;
      if (RAGGED && row >= nrows) break;
      float v = fmaxf(c[r] + bv, 0.0f);
      if (READOUT)
        atomicAdd(&gout[(size_t)nid[row] * KOUT + col], v);
      else
        ybase[(size_t)r * KOUT] = v;
    }
  }
}

// SAGE layer: Y = relu(Hs @ Ws^T + Hn @ Wn^T + bias); KIN=32, KOUT=64.
template <bool RAGGED>
__global__ void sage_layer(const float* __restrict__ Hs,
                           const float* __restrict__ Hn,
                           const float* __restrict__ Ws,
                           const float* __restrict__ Wn,
                           const float* __restrict__ bias,
                           float* __restrict__ Y, int nrows) {
  int r0 = scalar_wave_id() * 16;
  if (r0 >= nrows) return;
  int lane = threadIdx.x & 31;
  int sub = lane & 15;
  int hi = lane >> 4;
  int rowA = r0 + sub;
  if (RAGGED && rowA >= nrows) rowA = nrows - 1;

  v16h as = load_a(Hs, rowA, 32, hi, 0, 1.0f);
  v16h an = load_a(Hn, rowA, 32, hi, 0, 1.0f);

#pragma unroll
  for (int ot = 0; ot < 4; ++ot) {
    v8f c = {};
    v16h bs = load_b(Ws, ot * 16, sub, 32, hi, 0);
    c = wmma_f16(as, bs, c);
    v16h bn = load_b(Wn, ot * 16, sub, 32, hi, 0);
    c = wmma_f16(an, bn, c);
    int col = ot * 16 + sub;
    float bv = bias[col];
    float* ybase = Y + (size_t)(r0 + 8 * hi) * 64 + col;
#pragma unroll
    for (int r = 0; r < 8; ++r) {
      if (RAGGED && (r0 + r + 8 * hi) >= nrows) break;
      ybase[(size_t)r * 64] = fmaxf(c[r] + bv, 0.0f);
    }
  }
}

// ---------------------------------------------------------------------------
// Irregular kernels (memory / atomic bound; indexed arrays fit in 192MB L2).
// ---------------------------------------------------------------------------
__global__ void zero_kernel4(float4* __restrict__ p, size_t n4) {
  size_t i = blockIdx.x * (size_t)blockDim.x + threadIdx.x;
  size_t stride = (size_t)gridDim.x * blockDim.x;
  float4 z = {0.0f, 0.0f, 0.0f, 0.0f};
  for (; i < n4; i += stride) p[i] = z;
}

__global__ void deg_kernel(const int* __restrict__ src, const int* __restrict__ dst,
                           float* __restrict__ dout, float* __restrict__ din, int E) {
  int e = (int)(blockIdx.x * (size_t)blockDim.x + threadIdx.x);
  if (e >= E) return;
  atomicAdd(&dout[src[e]], 1.0f);
  atomicAdd(&din[dst[e]], 1.0f);
}

// h16[n][f] = embed[tokens[n]][f] * rsqrt(max(deg_out[n],1))
__global__ void embed_kernel(const int* __restrict__ tok,
                             const float* __restrict__ emb,
                             const float* __restrict__ degOut,
                             float* __restrict__ h16, int N) {
  size_t tid = blockIdx.x * (size_t)blockDim.x + threadIdx.x;
  if (tid >= (size_t)N * 16) return;
  int n = (int)(tid >> 4), f = (int)(tid & 15);
  float s = rsqrtf(fmaxf(degOut[n], 1.0f));
  h16[tid] = emb[(size_t)tok[n] * 16 + f] * s;
}

// m16[dst[e]][f] += h16[src[e]][f]
__global__ void scatter_add16(const int* __restrict__ src, const int* __restrict__ dst,
                              const float* __restrict__ h16, float* __restrict__ m16,
                              int E) {
  size_t tid = blockIdx.x * (size_t)blockDim.x + threadIdx.x;
  if (tid >= (size_t)E * 16) return;
  int e = (int)(tid >> 4), f = (int)(tid & 15);
  atomicAdd(&m16[(size_t)dst[e] * 16 + f], h16[(size_t)src[e] * 16 + f]);
}

// neigh[dst[e]][f] = max(neigh, hp[src[e]][f]); hp >= 0 (relu) so unsigned-bit
// atomicMax over a 0-initialized buffer is exact and implements the
// "empty neighborhood -> 0" clause for free.
__global__ void scatter_max32(const int* __restrict__ src, const int* __restrict__ dst,
                              const float* __restrict__ hp, float* __restrict__ neigh,
                              int E) {
  size_t tid = blockIdx.x * (size_t)blockDim.x + threadIdx.x;
  if (tid >= (size_t)E * 32) return;
  int e = (int)(tid >> 5), f = (int)(tid & 31);
  float v = hp[(size_t)src[e] * 32 + f];
  atomicMax((unsigned int*)&neigh[(size_t)dst[e] * 32 + f], __float_as_uint(v));
}

__global__ void cosine_kernel(const float* __restrict__ b1,
                              const float* __restrict__ b2,
                              float* __restrict__ out, int G) {
  int g = (int)(blockIdx.x * (size_t)blockDim.x + threadIdx.x);
  if (g >= G) return;
  float dot = 0.0f, n1 = 0.0f, n2 = 0.0f;
#pragma unroll 4
  for (int i = 0; i < 128; ++i) {
    float x = b1[(size_t)g * 128 + i];
    float y = b2[(size_t)g * 128 + i];
    dot += x * y; n1 += x * x; n2 += y * y;
  }
  float d1 = fmaxf(sqrtf(n1), 1e-8f);
  float d2 = fmaxf(sqrtf(n2), 1e-8f);
  out[g] = fminf(fabsf(dot / (d1 * d2)), 1.0f);
}

// ---------------------------------------------------------------------------
extern "C" void kernel_launch(void* const* d_in, const int* in_sizes, int n_in,
                              void* d_out, int out_size, void* d_ws, size_t ws_size,
                              hipStream_t stream) {
  const int N = in_sizes[0];   // 100000
  const int E = in_sizes[1];   // 3200000
  const int G = out_size;      // 512

  const int* tok[2] = {(const int*)d_in[0], (const int*)d_in[4]};
  const int* src[2] = {(const int*)d_in[1], (const int*)d_in[5]};
  const int* dst[2] = {(const int*)d_in[2], (const int*)d_in[6]};
  const int* nid[2] = {(const int*)d_in[3], (const int*)d_in[7]};
  const float* embed   = (const float*)d_in[8];
  const float* gcn1_W  = (const float*)d_in[9];
  const float* gcn1_b  = (const float*)d_in[10];
  const float* pool_W  = (const float*)d_in[11];
  const float* pool_b  = (const float*)d_in[12];
  const float* self_W  = (const float*)d_in[13];
  const float* neigh_W = (const float*)d_in[14];
  const float* sage_b  = (const float*)d_in[15];
  const float* lg_W    = (const float*)d_in[16];
  const float* lg_b    = (const float*)d_in[17];
  const float* top_W   = (const float*)d_in[18];
  const float* top_b   = (const float*)d_in[19];
  float* out = (float*)d_out;

  // Workspace layout (floats). Zero-init region is contiguous at the front.
  float* ws = (float*)d_ws;
  size_t off = 0;
  float* deg_out = ws + off; off += (size_t)N;
  float* deg_in  = ws + off; off += (size_t)N;
  float* m16     = ws + off; off += (size_t)N * 16;
  float* neigh   = ws + off; off += (size_t)N * 32;
  float* g64     = ws + off; off += (size_t)G * 64;
  size_t zero_count = off;                       // zero per encoder pass
  float* h16 = ws + off; off += (size_t)N * 16;
  float* h32 = ws + off; off += (size_t)N * 32;
  float* hp  = ws + off; off += (size_t)N * 32;
  float* h64 = ws + off; off += (size_t)N * 64;
  float* b1  = ws + off; off += (size_t)G * 128;
  float* b2  = ws + off; off += (size_t)G * 128;
  float* bmat[2] = {b1, b2};
  (void)ws_size; (void)n_in;

  const int T = 256;
  int tilesN = (N + 15) / 16;
  int blkDenseN = (tilesN * 32 + T - 1) / T;
  int tilesG = (G + 15) / 16;
  int blkDenseG = (tilesG * 32 + T - 1) / T;
  size_t zero4 = (zero_count + 3) / 4;  // components are multiples of 4 floats
  const bool ragN = (N & 15) != 0;      // false for N=100000
  const bool ragG = (G & 15) != 0;      // false for G=512

  for (int enc = 0; enc < 2; ++enc) {
    zero_kernel4<<<2048, T, 0, stream>>>((float4*)ws, zero4);
    deg_kernel<<<(E + T - 1) / T, T, 0, stream>>>(src[enc], dst[enc],
                                                  deg_out, deg_in, E);
    embed_kernel<<<((size_t)N * 16 + T - 1) / T, T, 0, stream>>>(
        tok[enc], embed, deg_out, h16, N);
    scatter_add16<<<((size_t)E * 16 + T - 1) / T, T, 0, stream>>>(
        src[enc], dst[enc], h16, m16, E);
    // GCN: h32 = relu((D_in^-1/2 * m16) @ gcn1_W^T + b)     K=16 (padded)
    if (ragN)
      dense_layer<16, 32, true, false, true><<<blkDenseN, T, 0, stream>>>(
          m16, gcn1_W, gcn1_b, h32, deg_in, nullptr, nullptr, N);
    else
      dense_layer<16, 32, true, false, false><<<blkDenseN, T, 0, stream>>>(
          m16, gcn1_W, gcn1_b, h32, deg_in, nullptr, nullptr, N);
    // pool: hp = relu(h32 @ pool_W^T + b)                   K=32
    if (ragN)
      dense_layer<32, 32, false, false, true><<<blkDenseN, T, 0, stream>>>(
          h32, pool_W, pool_b, hp, nullptr, nullptr, nullptr, N);
    else
      dense_layer<32, 32, false, false, false><<<blkDenseN, T, 0, stream>>>(
          h32, pool_W, pool_b, hp, nullptr, nullptr, nullptr, N);
    scatter_max32<<<((size_t)E * 32 + T - 1) / T, T, 0, stream>>>(
        src[enc], dst[enc], hp, neigh, E);
    // SAGE: h64 = relu(h32 @ self_W^T + neigh @ neigh_W^T + b)
    if (ragN)
      sage_layer<true><<<blkDenseN, T, 0, stream>>>(h32, neigh, self_W,
                                                    neigh_W, sage_b, h64, N);
    else
      sage_layer<false><<<blkDenseN, T, 0, stream>>>(h32, neigh, self_W,
                                                     neigh_W, sage_b, h64, N);
    // lg + fused segment-sum readout into g64                K=64 (2 WMMA)
    if (ragN)
      dense_layer<64, 64, false, true, true><<<blkDenseN, T, 0, stream>>>(
          h64, lg_W, lg_b, nullptr, nullptr, nid[enc], g64, N);
    else
      dense_layer<64, 64, false, true, false><<<blkDenseN, T, 0, stream>>>(
          h64, lg_W, lg_b, nullptr, nullptr, nid[enc], g64, N);
    // top: bmat = relu(g64 @ top_W^T + b)                    K=64, 128 out
    if (ragG)
      dense_layer<64, 128, false, false, true><<<blkDenseG, T, 0, stream>>>(
          g64, top_W, top_b, bmat[enc], nullptr, nullptr, nullptr, G);
    else
      dense_layer<64, 128, false, false, false><<<blkDenseG, T, 0, stream>>>(
          g64, top_W, top_b, bmat[enc], nullptr, nullptr, nullptr, G);
  }
  cosine_kernel<<<(G + T - 1) / T, T, 0, stream>>>(b1, b2, out, G);
}
